// RotaryMultiheadAttention_6691559047738
// MI455X (gfx1250) — matrix operand order repaired and checked by reading, not compiled, not run on hardware
//
#include <hip/hip_runtime.h>
#include <hip/hip_bf16.h>

// ---------------------------------------------------------------------------
// RotaryMultiheadAttention forward for gfx1250 (MI455X), wave32 + WMMA.
// S=2048, B=2, D=1024, H=16, HD=64.  d_out = out(S*B*D) ++ attn(B*H*S*S), fp32.
//
// Roofline: attn matrix is 537 MB fp32 and must be emitted once (reference
// returns it) -> >= 23 us at 23.3 TB/s.  scores+softmax+ctx are fused so
// attn is written exactly once and never re-read: a 512-thread workgroup
// owns a 16-row strip of one (b,h), keeps all 128 score tiles of that strip
// in registers across 16 waves, reduces the softmax statistics through LDS,
// and feeds the normalized probabilities (via an LDS re-layout to WMMA
// A-fragments) straight back into WMMA against V.  All matrix math is
// v_wmma_f32_16x16x32_f16 (f32 accumulate).
//
// Cache control: the 537 MB attn stream and the 16 MB final output are
// written with non-temporal hints (th:TH_STORE_NT) so they do not sweep the
// 192 MB L2; the f16 Q/K/V (24 MB) and ctx stay L2-resident for re-use.
//
// d_ws layout: qh/kh/vh f16 (B,H,S,HD) 8MB each, then ctx fp32 (B,S,D) 16MB.
// ---------------------------------------------------------------------------

typedef __attribute__((ext_vector_type(16))) _Float16 v16h;
typedef __attribute__((ext_vector_type(8)))  float    v8f;

#define S_LEN  2048
#define BATCH  2
#define DMODEL 1024
#define NHEAD  16
#define HDIM   64
#define BHEADS (BATCH * NHEAD)
#define QKV_ELEMS ((size_t)BATCH * NHEAD * S_LEN * HDIM)   // 4,194,304 halves

#define NT_WAVES 16                       // waves per fused block
#define TPW (S_LEN / 16 / NT_WAVES)       // score tiles per wave = 8

// ln(10000)/32 : RoPE inv_freq[f] = exp(-f * LOG_BASE_OVER_32)
#define LOG_BASE_OVER_32 0.28782313662425576f

// WMMA wave32 fragment maps (CDNA5 ISA):
// A (16-bit 16x32): lane row = lane&15; e=0..7 -> K=(lane<16?0:8)+e,
//                   e=8..15 -> K=(lane<16?16:24)+(e-8).
// B (16-bit 32x16): lane col = lane&15; K map as A; this file loads A and B both as K=(lane<16?0:16)+e.
// C/D (f32 16x16):  VGPR r, lane l -> M = r + (l<16?0:8), N = l&15.

__device__ __forceinline__ v8f wmma_f16(const v16h& a, const v16h& b, const v8f& c) {
    return __builtin_amdgcn_wmma_f32_16x16x32_f16(false, a, false, b, (short)0, c,
                                                  false, false);
}

// ---------------------------------------------------------------------------
// Kernel 1: y = x @ W^T + bias, optional fused RoPE, f16 out in (b,h,s,hd).
// grid = (D/16, S*B/16), block = 32 (one wave per 16x16 tile).
// ---------------------------------------------------------------------------
__global__ void __launch_bounds__(32)
qkv_proj_rope(const float* __restrict__ x, const float* __restrict__ W,
              const float* __restrict__ bias, _Float16* __restrict__ dst,
              int do_rope)
{
    const int lane = threadIdx.x;
    const int tn   = blockIdx.x;
    const int tm   = blockIdx.y;
    const int col  = tn * 16 + (lane & 15);
    const int arow = tm * 16 + (lane & 15);

    const int aoff0 = (lane < 16) ? 0 : 16;
    const int aoff1 = (lane < 16) ? 8 : 24;
    const int boff  = (lane < 16) ? 0 : 16;

    const float* __restrict__ arow_p = x + (size_t)arow * DMODEL;
    const float* __restrict__ bcol_p = W + (size_t)col * DMODEL;

    v8f acc = {};
    for (int k0 = 0; k0 < DMODEL; k0 += 32) {
        __builtin_prefetch(arow_p + k0 + 64, 0, 1);
        __builtin_prefetch(bcol_p + k0 + 64, 0, 1);
        v16h a, b;
        const float* a0 = arow_p + k0 + aoff0;
        const float* a1 = arow_p + k0 + aoff1;
        const float* bp = bcol_p + k0 + boff;
#pragma unroll
        for (int e = 0; e < 8; ++e) {
            a[e]     = (_Float16)a0[e];
            a[8 + e] = (_Float16)a1[e];
        }
#pragma unroll
        for (int e = 0; e < 16; ++e) b[e] = (_Float16)bp[e];
        acc = wmma_f16(a, b, acc);
    }

    const float bval = bias[col];
    const int h  = col / HDIM;
    const int hd = col % HDIM;
    const float inv  = __expf(-(float)(hd & 30) * LOG_BASE_OVER_32);
    const int   rbase = tm * 16 + ((lane >= 16) ? 8 : 0);

#pragma unroll
    for (int r = 0; r < 8; ++r) {
        const int m  = rbase + r;
        const int s  = m >> 1;          // BATCH == 2
        const int bb = m & 1;
        float v = acc[r] + bval;
        if (do_rope) {
            const float partner = __shfl_xor(v, 1, 32);
            const float ang = (float)s * inv;
            const float c = __cosf(ang);
            const float sn = __sinf(ang);
            v = (hd & 1) ? (partner * sn + v * c)
                         : (v * c - partner * sn);
        }
        // re-read soon by fused_attn -> keep regular temporal hint
        dst[((size_t)(bb * NHEAD + h) * S_LEN + s) * HDIM + hd] = (_Float16)v;
    }
}

// ---------------------------------------------------------------------------
// Kernel 2 (fused): scores -> softmax -> attn store (once, NT) -> ctx, one
// workgroup per (16-row strip, bh).  grid = (S/16, B*H), block = 512.
// ---------------------------------------------------------------------------
__global__ void __launch_bounds__(512)
fused_attn(const _Float16* __restrict__ qh, const _Float16* __restrict__ kh,
           const _Float16* __restrict__ vh, float* __restrict__ attn,
           float* __restrict__ ctx)
{
    __shared__ _Float16 stage[NT_WAVES][16][32];   // per-wave P re-layout tile
    __shared__ float    redbuf[NT_WAVES][16];      // row max / row sum
    __shared__ float    ctxsum[16][HDIM];          // cross-wave ctx reduction

    const int tid  = threadIdx.x;
    const int w    = tid >> 5;
    const int lane = tid & 31;
    const int tm   = blockIdx.x;
    const int bh   = blockIdx.y;
    const int b    = bh / NHEAD;
    const int h    = bh % NHEAD;

    const _Float16* __restrict__ qb = qh + (size_t)bh * S_LEN * HDIM;
    const _Float16* __restrict__ kb = kh + (size_t)bh * S_LEN * HDIM;
    const _Float16* __restrict__ vb = vh + (size_t)bh * S_LEN * HDIM;

    const int half  = (lane >= 16);
    const int l15   = lane & 15;
    const int aoff0 = half ? 16 : 0;
    const int aoff1 = half ? 24 : 8;
    const int rowof = half ? 8  : 0;

    // init ctx reduction buffer
    for (int i = tid; i < 16 * HDIM; i += 512) ((float*)ctxsum)[i] = 0.f;

    // ---- load Q A-fragments for this strip (shared by all its tiles) ----
    const int srow = tm * 16 + l15;
    v16h aq[2];
#pragma unroll
    for (int ks = 0; ks < 2; ++ks) {
        const _Float16* a0 = qb + (size_t)srow * HDIM + ks * 32 + aoff0;
        const _Float16* a1 = qb + (size_t)srow * HDIM + ks * 32 + aoff1;
#pragma unroll
        for (int e = 0; e < 8; ++e) { aq[ks][e] = a0[e]; aq[ks][8 + e] = a1[e]; }
    }

    // ---- Phase 1: raw scores for this wave's 8 tiles, kept in VGPRs ----
    v8f acc[TPW];
#pragma unroll
    for (int j = 0; j < TPW; ++j) {
        const int tcol = (w + NT_WAVES * j) * 16 + l15;
        v8f s = {};
#pragma unroll
        for (int ks = 0; ks < 2; ++ks) {
            v16h bf;
            const _Float16* bp = kb + (size_t)tcol * HDIM + ks * 32 + (half ? 16 : 0);
#pragma unroll
            for (int e = 0; e < 16; ++e) bf[e] = bp[e];
            s = wmma_f16(aq[ks], bf, s);
        }
        acc[j] = s;
    }

    // ---- Phase 2: global row max (lane group -> wave -> block) ----
    float mx[8];
#pragma unroll
    for (int r = 0; r < 8; ++r) {
        float m = acc[0][r];
#pragma unroll
        for (int j = 1; j < TPW; ++j) m = fmaxf(m, acc[j][r]);
#pragma unroll
        for (int msk = 1; msk < 16; msk <<= 1) m = fmaxf(m, __shfl_xor(m, msk, 32));
        mx[r] = m;
    }
    if (l15 == 0) {
#pragma unroll
        for (int r = 0; r < 8; ++r) redbuf[w][rowof + r] = mx[r];
    }
    __syncthreads();
    float gmax[8];
#pragma unroll
    for (int r = 0; r < 8; ++r) {
        float m = -3.402823466e38f;
        for (int w2 = 0; w2 < NT_WAVES; ++w2) m = fmaxf(m, redbuf[w2][rowof + r]);
        gmax[r] = m * 0.125f;          // 1/sqrt(HD); max commutes with +scale
    }
    __syncthreads();

    // ---- Phase 3: exp and global row sum ----
    float sm[8];
#pragma unroll
    for (int r = 0; r < 8; ++r) sm[r] = 0.f;
#pragma unroll
    for (int j = 0; j < TPW; ++j) {
#pragma unroll
        for (int r = 0; r < 8; ++r) {
            const float p = __expf(acc[j][r] * 0.125f - gmax[r]);
            acc[j][r] = p;
            sm[r] += p;
        }
    }
#pragma unroll
    for (int r = 0; r < 8; ++r) {
#pragma unroll
        for (int msk = 1; msk < 16; msk <<= 1) sm[r] += __shfl_xor(sm[r], msk, 32);
    }
    if (l15 == 0) {
#pragma unroll
        for (int r = 0; r < 8; ++r) redbuf[w][rowof + r] = sm[r];
    }
    __syncthreads();
    float rinv[8];
#pragma unroll
    for (int r = 0; r < 8; ++r) {
        float s = 0.f;
        for (int w2 = 0; w2 < NT_WAVES; ++w2) s += redbuf[w2][rowof + r];
        rinv[r] = 1.0f / s;
    }

    // ---- Phase 4: normalize, single streaming (non-temporal) attn store ----
    float* __restrict__ outp = attn + (size_t)bh * S_LEN * S_LEN;
#pragma unroll
    for (int j = 0; j < TPW; ++j) {
        const int tcol = (w + NT_WAVES * j) * 16 + l15;
#pragma unroll
        for (int r = 0; r < 8; ++r) {
            const float p = acc[j][r] * rinv[r];
            acc[j][r] = p;
            __builtin_nontemporal_store(
                p, &outp[(size_t)(tm * 16 + rowof + r) * S_LEN + tcol]);
        }
    }

    // ---- Phase 5: ctx partial = P(strip) @ V via WMMA over tile pairs ----
    // D-layout P -> LDS -> A-fragment layout, per wave (wave-internal RAW
    // covered by explicit s_wait_dscnt; DS is in-order within a wave).
    v8f cacc[4] = {};
#pragma unroll
    for (int i = 0; i < TPW / 2; ++i) {
#pragma unroll
        for (int jj = 0; jj < 2; ++jj) {
            const int j = 2 * i + jj;
#pragma unroll
            for (int r = 0; r < 8; ++r)
                stage[w][rowof + r][jj * 16 + l15] = (_Float16)acc[j][r];
        }
        asm volatile("s_wait_dscnt 0x0" ::: "memory");
        v16h pa;
        {
            const _Float16* s0 = &stage[w][l15][aoff0];
            const _Float16* s1 = &stage[w][l15][aoff1];
#pragma unroll
            for (int e = 0; e < 8; ++e) { pa[e] = s0[e]; pa[8 + e] = s1[e]; }
        }
        const int t0 = (w + NT_WAVES * (2 * i)) * 16;
        const int t1 = (w + NT_WAVES * (2 * i + 1)) * 16;
        const int tb = half ? t1 : t0;   // B K-halves follow the A K-halves
#pragma unroll
        for (int n = 0; n < 4; ++n) {
            const int hd = n * 16 + l15;
            v16h bf;
#pragma unroll
            for (int e = 0; e < 16; ++e)
                bf[e] = vb[(size_t)(tb + e) * HDIM + hd];
            cacc[n] = wmma_f16(pa, bf, cacc[n]);
        }
    }

    // ---- Phase 6: cross-wave reduction of ctx via LDS float atomics ----
#pragma unroll
    for (int n = 0; n < 4; ++n) {
#pragma unroll
        for (int r = 0; r < 8; ++r)
            atomicAdd(&ctxsum[rowof + r][n * 16 + l15], cacc[n][r]);
    }
    __syncthreads();
    for (int i = tid; i < 16 * HDIM; i += 512) {
        const int row = i >> 6;
        const int col = i & (HDIM - 1);
        // ctx is re-read by out_proj -> keep regular temporal hint
        ctx[((size_t)b * S_LEN + tm * 16 + row) * DMODEL + h * HDIM + col] =
            ctxsum[row][col];
    }
}

// ---------------------------------------------------------------------------
// Kernel 3: out[s,b,e] = ctx[b,s,:] . Wo[e,:] + bo[e].
// grid = (D/16, B*S/16), block = 32.  Final output -> non-temporal stores.
// ---------------------------------------------------------------------------
__global__ void __launch_bounds__(32)
out_proj(const float* __restrict__ ctx, const float* __restrict__ Wo,
         const float* __restrict__ bo, float* __restrict__ out)
{
    const int lane = threadIdx.x;
    const int tn   = blockIdx.x;
    const int tm   = blockIdx.y;
    const int col  = tn * 16 + (lane & 15);
    const int arow = tm * 16 + (lane & 15);

    const int aoff0 = (lane < 16) ? 0 : 16;
    const int aoff1 = (lane < 16) ? 8 : 24;
    const int boff  = (lane < 16) ? 0 : 16;

    const float* __restrict__ arow_p = ctx + (size_t)arow * DMODEL;
    const float* __restrict__ bcol_p = Wo  + (size_t)col * DMODEL;

    v8f acc = {};
    for (int k0 = 0; k0 < DMODEL; k0 += 32) {
        __builtin_prefetch(arow_p + k0 + 64, 0, 1);
        __builtin_prefetch(bcol_p + k0 + 64, 0, 1);
        v16h a, b;
        const float* a0 = arow_p + k0 + aoff0;
        const float* a1 = arow_p + k0 + aoff1;
        const float* bp = bcol_p + k0 + boff;
#pragma unroll
        for (int e = 0; e < 8; ++e) {
            a[e]     = (_Float16)a0[e];
            a[8 + e] = (_Float16)a1[e];
        }
#pragma unroll
        for (int e = 0; e < 16; ++e) b[e] = (_Float16)bp[e];
        acc = wmma_f16(a, b, acc);
    }

    const float bval = bo[col];
    const int rbase = tm * 16 + ((lane >= 16) ? 8 : 0);
#pragma unroll
    for (int r = 0; r < 8; ++r) {
        const int m = rbase + r;           // m = b*S + s
        const int bb = m / S_LEN;
        const int s  = m % S_LEN;
        __builtin_nontemporal_store(
            acc[r] + bval, &out[((size_t)s * BATCH + bb) * DMODEL + col]);
    }
}

// ---------------------------------------------------------------------------
extern "C" void kernel_launch(void* const* d_in, const int* in_sizes, int n_in,
                              void* d_out, int out_size, void* d_ws, size_t ws_size,
                              hipStream_t stream) {
    const float* query = (const float*)d_in[0];
    const float* key   = (const float*)d_in[1];
    const float* value = (const float*)d_in[2];
    const float* Wq = (const float*)d_in[3];
    const float* bq = (const float*)d_in[4];
    const float* Wk = (const float*)d_in[5];
    const float* bk = (const float*)d_in[6];
    const float* Wv = (const float*)d_in[7];
    const float* bv = (const float*)d_in[8];
    const float* Wo = (const float*)d_in[9];
    const float* bo = (const float*)d_in[10];

    float* out  = (float*)d_out;
    float* attn = out + (size_t)S_LEN * BATCH * DMODEL;

    _Float16* qh = (_Float16*)d_ws;
    _Float16* kh = qh + QKV_ELEMS;
    _Float16* vh = kh + QKV_ELEMS;
    float*    ctx = (float*)(vh + QKV_ELEMS);

    const dim3 b32(32);
    const dim3 gProj(DMODEL / 16, (S_LEN * BATCH) / 16);
    qkv_proj_rope<<<gProj, b32, 0, stream>>>(query, Wq, bq, qh, 1);
    qkv_proj_rope<<<gProj, b32, 0, stream>>>(key,   Wk, bk, kh, 1);
    qkv_proj_rope<<<gProj, b32, 0, stream>>>(value, Wv, bv, vh, 0);

    fused_attn<<<dim3(S_LEN / 16, BHEADS), dim3(512), 0, stream>>>(qh, kh, vh,
                                                                   attn, ctx);

    out_proj<<<gProj, b32, 0, stream>>>(ctx, Wo, bo, out);
}
